// ReadoutFunction_27178553049753
// MI455X (gfx1250) — matrix operand order repaired and checked by reading, not compiled, not run on hardware
//
#include <hip/hip_runtime.h>

typedef __attribute__((ext_vector_type(16))) _Float16 v16h;
typedef __attribute__((ext_vector_type(8)))  _Float16 v8h;
typedef __attribute__((ext_vector_type(8)))  float    v8f;

#define D 128
#define LDS_STRIDE 136   // 128 + 8 halves of padding -> 272B row stride, avoids bank conflicts

// ---------------------------------------------------------------- zero accum
__global__ __launch_bounds__(128) void duv_zero(float* __restrict__ acc) {
    acc[threadIdx.x] = 0.0f;
}

// ---------------------------------------------------------------- main: WMMA softmax-readout
__global__ __launch_bounds__(256) void duv_main(
    const float* __restrict__ h0, const float* __restrict__ h1,
    const float* __restrict__ h2, const float* __restrict__ h3,
    const float* __restrict__ W0, const float* __restrict__ W1,
    const float* __restrict__ W2, const float* __restrict__ W3,
    float* __restrict__ acc, int nstrips)
{
    __shared__ _Float16 Wt[D * LDS_STRIDE];   // W transposed: Wt[n][k]
    __shared__ float    wg_acc[D];

    const float* h; const float* W;
    switch (blockIdx.y) {
        case 0:  h = h0; W = W0; break;
        case 1:  h = h1; W = W1; break;
        case 2:  h = h2; W = W2; break;
        default: h = h3; W = W3; break;
    }

    const int tid = threadIdx.x;

    // Stage W into LDS, converted to f16, transposed to n-major.
    for (int i = tid; i < D * D; i += 256) {
        const int k = i >> 7;
        const int n = i & 127;
        Wt[n * LDS_STRIDE + k] = (_Float16)W[i];   // W[i] == W[k][n]
    }
    if (tid < D) wg_acc[tid] = 0.0f;
    __syncthreads();

    const int lane  = tid & 31;
    const int m     = lane & 15;       // row-in-strip for A, col-in-tile for B
    const int hi    = lane >> 4;       // lane half
    const int koffA = hi * 8;          // A layout: lane halves split K by 8
    const int koffB = hi * 8;          // B layout: lane halves split K by 8
    const int wid   = blockIdx.x * 8 + (tid >> 5);
    const int nw    = gridDim.x * 8;

    const v8f vzero = {0.f, 0.f, 0.f, 0.f, 0.f, 0.f, 0.f, 0.f};

    float colacc[8];
    #pragma unroll
    for (int t = 0; t < 8; ++t) colacc[t] = 0.0f;

    // ---- software-pipelined raw A data (fp32): 16 x float4 = one 16-row strip
    float4 f[8];   // first float4 of each 8-float chunk
    float4 g[8];   // second float4 of each 8-float chunk
    if (wid < nstrips) {
        const float* hrow = h + (size_t)(wid * 16 + m) * D;
        #pragma unroll
        for (int ks = 0; ks < 4; ++ks) {
            const int kb = ks * 32 + koffA;
            const float4* p0 = reinterpret_cast<const float4*>(hrow + kb);
            const float4* p1 = reinterpret_cast<const float4*>(hrow + kb + 16);
            f[2 * ks]     = p0[0];
            g[2 * ks]     = p0[1];
            f[2 * ks + 1] = p1[0];
            g[2 * ks + 1] = p1[1];
        }
    }

    for (int strip = wid; strip < nstrips; strip += nw) {
        // ---- convert the in-flight raw data into A fragments (CDNA5 f16 A layout)
        v16h a[4];
        #pragma unroll
        for (int ks = 0; ks < 4; ++ks) {
            const float4 x0 = f[2 * ks],     x1 = g[2 * ks];
            const float4 y0 = f[2 * ks + 1], y1 = g[2 * ks + 1];
            a[ks][0]  = (_Float16)x0.x; a[ks][1]  = (_Float16)x0.y;
            a[ks][2]  = (_Float16)x0.z; a[ks][3]  = (_Float16)x0.w;
            a[ks][4]  = (_Float16)x1.x; a[ks][5]  = (_Float16)x1.y;
            a[ks][6]  = (_Float16)x1.z; a[ks][7]  = (_Float16)x1.w;
            a[ks][8]  = (_Float16)y0.x; a[ks][9]  = (_Float16)y0.y;
            a[ks][10] = (_Float16)y0.z; a[ks][11] = (_Float16)y0.w;
            a[ks][12] = (_Float16)y1.x; a[ks][13] = (_Float16)y1.y;
            a[ks][14] = (_Float16)y1.z; a[ks][15] = (_Float16)y1.w;
        }

        // ---- issue next strip's global loads now; they stay in flight across
        //      the whole WMMA + softmax phase below.
        if (strip + nw < nstrips) {
            const float* hnext = h + (size_t)((strip + nw) * 16 + m) * D;
            #pragma unroll
            for (int ks = 0; ks < 4; ++ks) {
                const int kb = ks * 32 + koffA;
                const float4* p0 = reinterpret_cast<const float4*>(hnext + kb);
                const float4* p1 = reinterpret_cast<const float4*>(hnext + kb + 16);
                f[2 * ks]     = p0[0];
                g[2 * ks]     = p0[1];
                f[2 * ks + 1] = p1[0];
                g[2 * ks + 1] = p1[1];
            }
        }

        // ---- opaque ZERO byte-offset: the address-space of Wt survives the
        //      GEP (so these stay ds_load_b128), but the loads become
        //      loop-variant, so LICM cannot hoist 256 VGPRs of B fragments
        //      out of the strip loop.
        unsigned wo = 0;
        asm volatile("" : "+v"(wo));
        const _Float16* wb =
            reinterpret_cast<const _Float16*>(reinterpret_cast<const char*>(&Wt[0]) + wo);

        // ---- 8 output tiles of 16x16, K accumulated in 4 steps of 32
        v8f c[8];
        #pragma unroll
        for (int t = 0; t < 8; ++t) c[t] = vzero;

        #pragma unroll
        for (int t = 0; t < 8; ++t) {
            const _Float16* wcol = wb + (t * 16 + m) * LDS_STRIDE;
            #pragma unroll
            for (int ks = 0; ks < 4; ++ks) {
                const int kb = ks * 32 + koffB;
                const v8h* pb = reinterpret_cast<const v8h*>(wcol + kb);
                const v8h blo = pb[0];
                const v8h bhi = pb[2];
                v16h b;
                #pragma unroll
                for (int j = 0; j < 8; ++j) { b[j] = blo[j]; b[8 + j] = bhi[j]; }
                c[t] = __builtin_amdgcn_wmma_f32_16x16x32_f16(
                        false, a[ks], false, b, (short)0, c[t], false, false);
            }
        }

        // ---- softmax over the 128 columns of each of the 16 rows, then
        //      accumulate probability column-sums.
        // C layout: VGPR r of tile t holds (row r, cols t*16+lane) in lanes 0-15
        //           and (row r+8, cols t*16+lane-16) in lanes 16-31.
        #pragma unroll
        for (int r = 0; r < 8; ++r) {
            float mx = c[0][r];
            #pragma unroll
            for (int t = 1; t < 8; ++t) mx = fmaxf(mx, c[t][r]);
            mx = fmaxf(mx, __shfl_xor(mx, 1, 32));
            mx = fmaxf(mx, __shfl_xor(mx, 2, 32));
            mx = fmaxf(mx, __shfl_xor(mx, 4, 32));
            mx = fmaxf(mx, __shfl_xor(mx, 8, 32));

            float sm = 0.0f;
            #pragma unroll
            for (int t = 0; t < 8; ++t) {
                const float e = __expf(c[t][r] - mx);
                c[t][r] = e;
                sm += e;
            }
            sm += __shfl_xor(sm, 1, 32);
            sm += __shfl_xor(sm, 2, 32);
            sm += __shfl_xor(sm, 4, 32);
            sm += __shfl_xor(sm, 8, 32);

            const float inv = 1.0f / sm;
            #pragma unroll
            for (int t = 0; t < 8; ++t)
                colacc[t] = __builtin_fmaf(c[t][r], inv, colacc[t]);
        }
    }

    // ---- combine lane halves (rows 0-7 + rows 8-15), reduce across waves
    #pragma unroll
    for (int t = 0; t < 8; ++t) {
        const float v = colacc[t] + __shfl_xor(colacc[t], 16, 32);
        if (lane < 16) atomicAdd(&wg_acc[t * 16 + lane], v);
    }
    __syncthreads();
    if (tid < D) atomicAdd(&acc[tid], wg_acc[tid]);
}

// ---------------------------------------------------------------- scalar tail (N % 16 rows; dead for N=100000)
__global__ __launch_bounds__(128) void duv_tail(
    const float* __restrict__ h0, const float* __restrict__ h1,
    const float* __restrict__ h2, const float* __restrict__ h3,
    const float* __restrict__ W0, const float* __restrict__ W1,
    const float* __restrict__ W2, const float* __restrict__ W3,
    float* __restrict__ acc, int row0, int N)
{
    const float* h; const float* W;
    switch (blockIdx.x) {
        case 0:  h = h0; W = W0; break;
        case 1:  h = h1; W = W1; break;
        case 2:  h = h2; W = W2; break;
        default: h = h3; W = W3; break;
    }
    __shared__ float logit[D];
    __shared__ float red[2];
    const int c = threadIdx.x;
    for (int row = row0; row < N; ++row) {
        float s = 0.0f;
        for (int k = 0; k < D; ++k) s += h[(size_t)row * D + k] * W[k * D + c];
        logit[c] = s;
        __syncthreads();
        if (c == 0) {
            float mx = logit[0];
            for (int i = 1; i < D; ++i) mx = fmaxf(mx, logit[i]);
            float sm = 0.0f;
            for (int i = 0; i < D; ++i) sm += __expf(logit[i] - mx);
            red[0] = mx; red[1] = sm;
        }
        __syncthreads();
        atomicAdd(&acc[c], __expf(logit[c] - red[0]) / red[1]);
        __syncthreads();
    }
}

// ---------------------------------------------------------------- final linear 128 -> 16
__global__ __launch_bounds__(32) void duv_final(
    const float* __restrict__ acc, const float* __restrict__ Wlin,
    const float* __restrict__ blin, float* __restrict__ out)
{
    const int t = threadIdx.x;
    if (t < 16) {
        float s = blin[t];
        for (int n = 0; n < D; ++n) s += acc[n] * Wlin[t * D + n];
        out[t] = s;
    }
}

extern "C" void kernel_launch(void* const* d_in, const int* in_sizes, int n_in,
                              void* d_out, int out_size, void* d_ws, size_t ws_size,
                              hipStream_t stream) {
    // dict order: h0, W0, h1, W1, h2, W2, h3, W3, Wlin, blin
    const float* h0   = (const float*)d_in[0];
    const float* W0   = (const float*)d_in[1];
    const float* h1   = (const float*)d_in[2];
    const float* W1   = (const float*)d_in[3];
    const float* h2   = (const float*)d_in[4];
    const float* W2   = (const float*)d_in[5];
    const float* h3   = (const float*)d_in[6];
    const float* W3   = (const float*)d_in[7];
    const float* Wlin = (const float*)d_in[8];
    const float* blin = (const float*)d_in[9];
    float* out = (float*)d_out;

    const int N       = in_sizes[0] / D;   // 100000
    const int nstrips = N / 16;            // 6250 (exact for 100000)
    const int rem     = N & 15;
    float* acc = (float*)d_ws;             // 128-float accumulator

    duv_zero<<<1, 128, 0, stream>>>(acc);

    dim3 grid(128, 4);                     // x: node-strip workgroups, y: layer
    duv_main<<<grid, 256, 0, stream>>>(h0, h1, h2, h3, W0, W1, W2, W3, acc, nstrips);

    if (rem) {
        duv_tail<<<4, 128, 0, stream>>>(h0, h1, h2, h3, W0, W1, W2, W3,
                                        acc, nstrips * 16, N);
    }

    duv_final<<<1, 32, 0, stream>>>(acc, Wlin, blin, out);
}
